// GNN_82171314307280
// MI455X (gfx1250) — matrix operand order repaired and checked by reading, not compiled, not run on hardware
//
#include <hip/hip_runtime.h>

typedef __attribute__((ext_vector_type(16))) _Float16 v16h;
typedef __attribute__((ext_vector_type(8)))  float    v8f;

union Frag16 {
    v16h v;
    unsigned int u[8];
};

// ---------------------------------------------------------------------------
// K0: zero scratch (cnt, aggpos, aggl, gsum) — graph-capture-safe memset
// ---------------------------------------------------------------------------
__global__ void k_zero(float4* __restrict__ p, long n4) {
    long i = (long)blockIdx.x * blockDim.x + threadIdx.x;
    if (i < n4) p[i] = make_float4(0.f, 0.f, 0.f, 0.f);
}

// ---------------------------------------------------------------------------
// Edge decode helper: edges (f0,f1), (f1,f2), (f0,f2); no integer division.
// ---------------------------------------------------------------------------
__device__ __forceinline__ void edge_nodes(const int* __restrict__ face, int e, int F,
                                           int& src, int& dst) {
    int which = (e >= F) + (e >= 2 * F);
    int f = e - which * F;
    if (which == 0)      { src = face[0 * F + f]; dst = face[1 * F + f]; }
    else if (which == 1) { src = face[1 * F + f]; dst = face[2 * F + f]; }
    else                 { src = face[0 * F + f]; dst = face[2 * F + f]; }
}

// ---------------------------------------------------------------------------
// K1: per-edge degree + position aggregation (layer-1 neighbor sum, d=3)
// ---------------------------------------------------------------------------
__global__ void k_edge_pos(const int* __restrict__ face, const float* __restrict__ pos,
                           float* __restrict__ cnt, float* __restrict__ aggpos, int F) {
    int e = blockIdx.x * blockDim.x + threadIdx.x;
    if (e >= 3 * F) return;
    int src, dst;
    edge_nodes(face, e, F, src, dst);
    if (src == dst) return;
    atomicAdd(&cnt[dst], 1.0f);
    atomicAdd(&aggpos[dst * 3 + 0], pos[src * 3 + 0]);
    atomicAdd(&aggpos[dst * 3 + 1], pos[src * 3 + 1]);
    atomicAdd(&aggpos[dst * 3 + 2], pos[src * 3 + 2]);
}

// ---------------------------------------------------------------------------
// K2: layer1: h1 = relu(mean_agg(pos) @ W1l + pos @ W1r + b1), stored f16.
// K=3 -> pure VALU is optimal; WMMA would be ~90% wasted lanes.
// ---------------------------------------------------------------------------
__global__ void k_layer1(const float* __restrict__ pos, const float* __restrict__ aggpos,
                         const float* __restrict__ cnt,
                         const float* __restrict__ W1l, const float* __restrict__ W1r,
                         const float* __restrict__ b1, _Float16* __restrict__ h1) {
    int n = blockIdx.x;
    int c = threadIdx.x;  // 256
    float inv = 1.0f / fmaxf(cnt[n], 1.0f);
    float a0 = aggpos[n * 3 + 0] * inv;
    float a1 = aggpos[n * 3 + 1] * inv;
    float a2 = aggpos[n * 3 + 2] * inv;
    float x0 = pos[n * 3 + 0], x1 = pos[n * 3 + 1], x2 = pos[n * 3 + 2];
    float s = b1[c];
    s = fmaf(a0, W1l[0 * 256 + c], s);
    s = fmaf(a1, W1l[1 * 256 + c], s);
    s = fmaf(a2, W1l[2 * 256 + c], s);
    s = fmaf(x0, W1r[0 * 256 + c], s);
    s = fmaf(x1, W1r[1 * 256 + c], s);
    s = fmaf(x2, W1r[2 * 256 + c], s);
    h1[(size_t)n * 256 + c] = (_Float16)fmaxf(s, 0.0f);
}

// ---------------------------------------------------------------------------
// K2b: pack W2 = [W2l | W2r] (256x256) to f16, column-major w2t[n][k]
// ---------------------------------------------------------------------------
__global__ void k_packW2(const float* __restrict__ W2l, const float* __restrict__ W2r,
                         _Float16* __restrict__ w2t) {
    int idx = blockIdx.x * blockDim.x + threadIdx.x;  // 65536
    int n = idx >> 8, k = idx & 255;
    float v = (n < 128) ? W2l[k * 128 + n] : W2r[k * 128 + (n - 128)];
    w2t[n * 256 + k] = (_Float16)v;
}

// ---------------------------------------------------------------------------
// K3: y[N,256] = h1f16[N,256] @ W2f16[256,256]  via v_wmma_f32_16x16x32_f16.
// One wave: 16 rows x 256 cols. A frags (K=0..255) cached in 64 VGPRs.
// Per column tile: preload ALL 8 B frags (clause of b128 loads, one wait),
// then 8 back-to-back WMMAs on the acc chain — no per-WMMA load stall.
// ISA 7.12.2 layouts: A lane l: m=l%16, g=l/16, v0..3 -> K=g*8+2v,
// v4..7 -> K=16+g*8+2(v-4). B (col-major src): lane col n=l%16,
// vgpr v -> same K as A. C/D: lane n=l%16, vgpr v -> row g*8+v.
// ---------------------------------------------------------------------------
__global__ void k_gemm_y(const _Float16* __restrict__ h1, const _Float16* __restrict__ w2t,
                         float* __restrict__ y, int N) {
    int wave = threadIdx.x >> 5;
    int lane = threadIdx.x & 31;
    int tile = blockIdx.x * 4 + wave;          // 16-row tile index
    int row0 = tile * 16;
    if (row0 >= N) return;
    int mrow = lane & 15;
    int g    = lane >> 4;
    int m = row0 + mrow;
    if (m >= N) m = N - 1;                     // duplicate-load clamp; stores guarded
    bool fullTile = (row0 + 16 <= N);

    const _Float16* arow = h1 + (size_t)m * 256;
    Frag16 af[8];
#pragma unroll
    for (int kc = 0; kc < 8; ++kc) {
        int kb = kc * 32;
#pragma unroll
        for (int v = 0; v < 4; ++v)
            af[kc].u[v] = *(const unsigned int*)(arow + kb + g * 8 + 2 * v);
#pragma unroll
        for (int v = 4; v < 8; ++v)
            af[kc].u[v] = *(const unsigned int*)(arow + kb + 16 + g * 8 + 2 * (v - 4));
    }

    for (int ct = 0; ct < 16; ++ct) {
        int ncol = ct * 16 + (lane & 15);
        const _Float16* bcol = w2t + (size_t)ncol * 256 + g * 8;

        // Preload all 8 B fragments for this column tile (64 VGPRs).
        Frag16 bf[8];
#pragma unroll
        for (int kc = 0; kc < 8; ++kc) {
#pragma unroll
            for (int v = 0; v < 8; ++v)
                bf[kc].u[v] = *(const unsigned int*)(bcol + kc * 32 + 2 * v + (v >= 4 ? 8 : 0));
        }

        // Dense WMMA accumulation chain.
        v8f acc = {};
#pragma unroll
        for (int kc = 0; kc < 8; ++kc)
            acc = __builtin_amdgcn_wmma_f32_16x16x32_f16(
                false, af[kc].v, false, bf[kc].v, (short)0, acc, false, false);

        // Store: lane holds col ncol; vgpr v -> row row0 + g*8 + v.
        float* yb = y + (size_t)(row0 + g * 8) * 256 + ncol;
        if (fullTile) {
#pragma unroll
            for (int v = 0; v < 8; ++v)
                yb[(size_t)v * 256] = acc[v];
        } else {
#pragma unroll
            for (int v = 0; v < 8; ++v)
                if (row0 + g * 8 + v < N) yb[(size_t)v * 256] = acc[v];
        }
    }
}

// ---------------------------------------------------------------------------
// K4: edge scatter of y_l = y[:, :128]: aggl[dst] += y_l[src]  (valid edges).
// 32 chunk-threads per edge, float4 loads, 4 f32 atomics each.
// ---------------------------------------------------------------------------
__global__ void k_edge_scatter(const int* __restrict__ face, const float* __restrict__ y,
                               float* __restrict__ aggl, int F) {
    int gid = blockIdx.x * blockDim.x + threadIdx.x;
    int e  = gid >> 5;
    int ch = gid & 31;
    if (e >= 3 * F) return;
    int src, dst;
    edge_nodes(face, e, F, src, dst);
    if (src == dst) return;
    const float4 v = *(const float4*)(y + (size_t)src * 256 + ch * 4);
    float* p = aggl + (size_t)dst * 128 + ch * 4;
    atomicAdd(p + 0, v.x);
    atomicAdd(p + 1, v.y);
    atomicAdd(p + 2, v.z);
    atomicAdd(p + 3, v.w);
}

// ---------------------------------------------------------------------------
// K5: h2 = relu(aggl/max(cnt,1) + y_r + b2); fused global mean pool (sum).
// ---------------------------------------------------------------------------
__global__ void k_pool(const float* __restrict__ aggl, const float* __restrict__ y,
                       const float* __restrict__ cnt, const float* __restrict__ b2,
                       float* __restrict__ gsum, int N) {
    int c  = threadIdx.x;  // 128
    int n0 = blockIdx.x * 64;
    float bc = b2[c];
    float s = 0.0f;
    for (int i = 0; i < 64; ++i) {
        int n = n0 + i;
        if (n >= N) break;
        float inv = 1.0f / fmaxf(cnt[n], 1.0f);
        float h = aggl[(size_t)n * 128 + c] * inv + y[(size_t)n * 256 + 128 + c] + bc;
        s += fmaxf(h, 0.0f);
    }
    atomicAdd(&gsum[c], s);
}

// ---------------------------------------------------------------------------
// K6: decoder: probs = softmax(mean(h2) @ Wd + bd); out[10] = argmax.
// ---------------------------------------------------------------------------
__global__ void k_decode(const float* __restrict__ gsum, const float* __restrict__ Wd,
                         const float* __restrict__ bd, float* __restrict__ out,
                         int out_size, int N) {
    if (threadIdx.x != 0 || blockIdx.x != 0) return;
    float invN = 1.0f / (float)N;
    float logits[10];
    float mx = -3.0e38f;
    for (int j = 0; j < 10; ++j) {
        float s = bd[j];
        for (int c = 0; c < 128; ++c) s = fmaf(gsum[c] * invN, Wd[c * 10 + j], s);
        logits[j] = s;
        mx = fmaxf(mx, s);
    }
    float denom = 0.0f;
    for (int j = 0; j < 10; ++j) { logits[j] = expf(logits[j] - mx); denom += logits[j]; }
    float inv = 1.0f / denom;
    int arg = 0; float best = -1.0f;
    for (int j = 0; j < 10; ++j) {
        float p = logits[j] * inv;
        out[j] = p;
        if (p > best) { best = p; arg = j; }
    }
    if (out_size > 10) out[10] = (float)arg;
}

// ---------------------------------------------------------------------------
extern "C" void kernel_launch(void* const* d_in, const int* in_sizes, int n_in,
                              void* d_out, int out_size, void* d_ws, size_t ws_size,
                              hipStream_t stream) {
    const float* pos  = (const float*)d_in[0];
    const int*   face = (const int*)  d_in[1];
    const float* W1l  = (const float*)d_in[2];
    const float* W1r  = (const float*)d_in[3];
    const float* b1   = (const float*)d_in[4];
    const float* W2l  = (const float*)d_in[5];
    const float* W2r  = (const float*)d_in[6];
    const float* b2   = (const float*)d_in[7];
    const float* Wd   = (const float*)d_in[8];
    const float* bd   = (const float*)d_in[9];

    const int N = in_sizes[0] / 3;   // 200000
    const int F = in_sizes[1] / 3;   // 400000
    const int E = 3 * F;

    // Workspace partition (256B aligned). Zeroed block laid first so one
    // kernel clears cnt+aggpos+aggl+gsum contiguously.
    char* ws = (char*)d_ws;
    size_t off = 0;
    auto alloc = [&](size_t bytes) -> void* {
        void* p = (void*)(ws + off);
        off += (bytes + 255) & ~(size_t)255;
        return p;
    };
    float*    cnt    = (float*)   alloc((size_t)N * sizeof(float));
    float*    aggpos = (float*)   alloc((size_t)N * 3 * sizeof(float));
    float*    aggl   = (float*)   alloc((size_t)N * 128 * sizeof(float));
    float*    gsum   = (float*)   alloc(128 * sizeof(float));
    size_t zero_bytes = off;
    _Float16* h1f16  = (_Float16*)alloc((size_t)N * 256 * sizeof(_Float16));
    float*    y      = (float*)   alloc((size_t)N * 256 * sizeof(float));
    _Float16* w2t    = (_Float16*)alloc(256 * 256 * sizeof(_Float16));

    float* out = (float*)d_out;

    long n4 = (long)(zero_bytes / 16);
    k_zero<<<(unsigned)((n4 + 255) / 256), 256, 0, stream>>>((float4*)ws, n4);

    k_edge_pos<<<(E + 255) / 256, 256, 0, stream>>>(face, pos, cnt, aggpos, F);

    k_layer1<<<N, 256, 0, stream>>>(pos, aggpos, cnt, W1l, W1r, b1, h1f16);

    k_packW2<<<256, 256, 0, stream>>>(W2l, W2r, w2t);

    int rowTiles = (N + 15) / 16;
    k_gemm_y<<<(rowTiles + 3) / 4, 128, 0, stream>>>(h1f16, w2t, y, N);

    long sthreads = (long)E * 32;
    k_edge_scatter<<<(unsigned)((sthreads + 255) / 256), 256, 0, stream>>>(face, y, aggl, F);

    k_pool<<<(N + 63) / 64, 128, 0, stream>>>(aggl, y, cnt, b2, gsum, N);

    k_decode<<<1, 32, 0, stream>>>(gsum, Wd, bd, out, out_size, N);
}
